// Codebook_9775345565922
// MI455X (gfx1250) — matrix operand order repaired and checked by reading, not compiled, not run on hardware
//
#include <hip/hip_runtime.h>
#include <hip/hip_bf16.h>
#include <float.h>
#include <stdint.h>

// ---------------------------------------------------------------------------
// VQ-VAE codebook quantization for MI455X (gfx1250, wave32, WMMA).
//   z [32,256,32,32] fp32, embedding [8192,256] fp32
// Outputs: z_q_out [8388608] fp32, min_d [32768] i32, loss [1] fp32.
//
// GEMM+argmin: split-bf16 (hi*hi + hi*lo + lo*hi) on v_wmma_f32_16x16x32_bf16.
// Each wave: 2 M-tiles register-resident (A hi/lo = 256 VGPRs).
// Each block: 8 waves share B tiles staged in LDS via async global->LDS
// copies (double buffered, ASYNCcnt). N split 4-way for occupancy; partial
// argmins merged in a final pass.
// ---------------------------------------------------------------------------

typedef __attribute__((ext_vector_type(16))) __bf16 v16bf;
typedef __attribute__((ext_vector_type(8)))  float  v8f;
typedef __attribute__((ext_vector_type(4)))  int    v4i;

#define BB    32
#define CC    256
#define HWW   1024
#define MM    (BB * HWW)      // 32768
#define KK    8192
#define CHW   (CC * HWW)      // 262144
#define TOTAL (BB * CHW)      // 8388608

#define NSPLIT   4
#define NT_PER   ((KK / 16) / NSPLIT)   // 128 N-tiles per block
#define TILE_B   8192                   // bytes of one bf16 plane tile (16 cols x 256)

// workspace layout (bytes)
#define EHI_OFF   0u
#define ELO_OFF   (KK * CC * 2u)                  // 4 MiB
#define ENORM_OFF (KK * CC * 4u)                  // 8 MiB
#define LOSS_OFF  (ENORM_OFF + KK * 4u)
#define PBEST_OFF (LOSS_OFF + 256u)
#define PIDX_OFF  (PBEST_OFF + (unsigned)(NSPLIT * MM * 4))
// total ~9.5 MB

#define WMMA_BF16(Ah, Bh, Cc) \
    __builtin_amdgcn_wmma_f32_16x16x32_bf16(false, (Ah), false, (Bh), (short)0, (Cc), false, false)

// ---- async global -> LDS copy of 16 bytes -------------------------------
__device__ __forceinline__ void async_copy16(const void* g, void* l) {
#if __has_builtin(__builtin_amdgcn_global_load_async_to_lds_b128)
    __builtin_amdgcn_global_load_async_to_lds_b128(
        (__attribute__((address_space(1))) v4i*)(v4i*)const_cast<void*>(g),
        (__attribute__((address_space(3))) v4i*)(v4i*)l,
        0, 0);
#else
    unsigned lds = (unsigned)(uintptr_t)l;           // low 32 bits = LDS offset
    unsigned long long ga = (unsigned long long)(uintptr_t)g;
    asm volatile("global_load_async_to_lds_b128 %0, %1, off"
                 :: "v"(lds), "v"(ga) : "memory");
#endif
}

#if __has_builtin(__builtin_amdgcn_s_wait_asynccnt)
#define WAIT_ASYNC(N) __builtin_amdgcn_s_wait_asynccnt(N)
#else
#define WAIT_ASYNC(N) asm volatile("s_wait_asynccnt %0" :: "i"(N) : "memory")
#endif

// ---------------------------------------------------------------------------
// Kernel 1: split embedding fp32 -> (hi, lo) bf16 planes; zero loss accum.
// ---------------------------------------------------------------------------
__global__ __launch_bounds__(256)
void vq_prep_split(const float* __restrict__ emb,
                   __bf16* __restrict__ ehi,
                   __bf16* __restrict__ elo,
                   float* __restrict__ loss_acc) {
    int i = blockIdx.x * blockDim.x + threadIdx.x;
    if (i == 0) *loss_acc = 0.0f;
    if (i < KK * CC) {
        float f = emb[i];
        __bf16 h = (__bf16)f;
        ehi[i] = h;
        elo[i] = (__bf16)(f - (float)h);
    }
}

// ---------------------------------------------------------------------------
// Kernel 2: ||e_k||^2 per codebook row (one wave per row).
// ---------------------------------------------------------------------------
__global__ __launch_bounds__(256)
void vq_prep_enorm(const float* __restrict__ emb,
                   float* __restrict__ enorm) {
    const int lane = threadIdx.x & 31;
    const int row  = blockIdx.x * 8 + (threadIdx.x >> 5);
    const float* p = emb + (size_t)row * CC;
    float s = 0.0f;
#pragma unroll
    for (int j = 0; j < CC / 32; ++j) {
        float v = p[lane + 32 * j];
        s += v * v;
    }
#pragma unroll
    for (int off = 16; off > 0; off >>= 1)
        s += __shfl_xor(s, off, 32);
    if (lane == 0) enorm[row] = s;
}

// ---------------------------------------------------------------------------
// Kernel 3: fused GEMM + argmin over an N chunk.
// Block: 8 waves x 2 M-tiles = 256 rows; loops over 128 N-tiles.
// B tiles double-buffered in LDS via async copies.
// ---------------------------------------------------------------------------
__global__ __launch_bounds__(256)
void vq_argmin(const float* __restrict__ z,
               const __bf16* __restrict__ ehi,
               const __bf16* __restrict__ elo,
               const float* __restrict__ enorm,
               float* __restrict__ pbest,
               int* __restrict__ pidx) {
    __shared__ __align__(32) char smem[2][2 * TILE_B];   // [buf][hi|lo]

    const int lane   = threadIdx.x & 31;
    const int wid    = threadIdx.x >> 5;
    const int mgroup = blockIdx.x & 127;          // 128 M groups of 256 rows
    const int nchunk = blockIdx.x >> 7;           // 4 N chunks
    const int ntBase = nchunk * NT_PER;
    const int rowBase = mgroup * 256 + wid * 32;  // 2 tiles: rows [base, base+32)

    // --- A fragments for both M tiles, register-resident (contiguous k order, same as B:
    // lanes<16 hold K=0..15, lanes>=16 hold K=16..31).
    const int asub = (lane & 16) ? 16 : 0;
    v16bf a0h[8], a0l[8], a1h[8], a1l[8];
#pragma unroll
    for (int t = 0; t < 2; ++t) {
        const float* zrow = z + (size_t)(rowBase + t * 16 + (lane & 15)) * CC;
#pragma unroll
        for (int q = 0; q < 8; ++q) {
            const float* p0 = zrow + 32 * q + asub;
            float f[16];
#pragma unroll
            for (int e = 0; e < 8; ++e) f[e] = p0[e];
#pragma unroll
            for (int e = 0; e < 8; ++e) f[8 + e] = p0[8 + e];
#pragma unroll
            for (int e = 0; e < 16; ++e) {
                __bf16 h = (__bf16)f[e];
                __bf16 l = (__bf16)(f[e] - (float)h);
                if (t == 0) { a0h[q][e] = h; a0l[q][e] = l; }
                else        { a1h[q][e] = h; a1l[q][e] = l; }
            }
        }
    }

    float best0[8], best1[8];
    int   bidx0[8], bidx1[8];
#pragma unroll
    for (int r = 0; r < 8; ++r) {
        best0[r] = FLT_MAX; best1[r] = FLT_MAX;
        bidx0[r] = 0;       bidx1[r] = 0;
    }

    const int bsub    = (lane & 16) ? 16 : 0;
    const int colLane = lane & 15;
    const int t4      = threadIdx.x * 16;         // staging offset per thread

    // stage first tile into buffer 0
    {
        const char* gh = (const char*)ehi + (size_t)ntBase * TILE_B;
        const char* gl = (const char*)elo + (size_t)ntBase * TILE_B;
#pragma unroll
        for (int r = 0; r < 2; ++r) {
            async_copy16(gh + t4 + r * 4096, smem[0] + t4 + r * 4096);
            async_copy16(gl + t4 + r * 4096, smem[0] + TILE_B + t4 + r * 4096);
        }
    }

    for (int i = 0; i < NT_PER; ++i) {
        const int nt = ntBase + i;
        char* cur = smem[i & 1];
        char* nxt = smem[(i + 1) & 1];

        if (i + 1 < NT_PER) {   // prefetch next tile, then wait for current
            const char* gh = (const char*)ehi + (size_t)(nt + 1) * TILE_B;
            const char* gl = (const char*)elo + (size_t)(nt + 1) * TILE_B;
#pragma unroll
            for (int r = 0; r < 2; ++r) {
                async_copy16(gh + t4 + r * 4096, nxt + t4 + r * 4096);
                async_copy16(gl + t4 + r * 4096, nxt + TILE_B + t4 + r * 4096);
            }
            WAIT_ASYNC(4);
        } else {
            WAIT_ASYNC(0);
        }
        __syncthreads();        // all waves' staging of `cur` complete

        // --- compute on cur: B 32x16 fragments from LDS (lane%16 = column;
        // lanes<16 hold K=0..15, lanes>=16 hold K=16..31 of each slab).
        const __bf16* lh = (const __bf16*)cur + colLane * 256 + bsub;
        const __bf16* ll = lh + TILE_B / 2;   // lo plane, +8192 bytes
        v8f acc0 = {}, acc1 = {};
#pragma unroll
        for (int q = 0; q < 8; ++q) {
            v16bf bh = *(const v16bf*)(lh + 32 * q);
            v16bf bl = *(const v16bf*)(ll + 32 * q);
            acc0 = WMMA_BF16(a0h[q], bh, acc0);
            acc1 = WMMA_BF16(a1h[q], bh, acc1);
            acc0 = WMMA_BF16(a0h[q], bl, acc0);
            acc1 = WMMA_BF16(a1h[q], bl, acc1);
            acc0 = WMMA_BF16(a0l[q], bh, acc0);
            acc1 = WMMA_BF16(a1l[q], bh, acc1);
        }
        const int   col = nt * 16 + colLane;
        const float en  = enorm[col];
#pragma unroll
        for (int r = 0; r < 8; ++r) {
            float s0 = en - 2.0f * acc0[r];
            float s1 = en - 2.0f * acc1[r];
            bool t0 = s0 < best0[r];
            bool t1 = s1 < best1[r];
            best0[r] = t0 ? s0 : best0[r];
            bidx0[r] = t0 ? col : bidx0[r];
            best1[r] = t1 ? s1 : best1[r];
            bidx1[r] = t1 ? col : bidx1[r];
        }
        __syncthreads();        // cur is free to be overwritten next round
    }

    // --- min-reduce across the 16 lanes of each half (C/D layout: VGPR r =
    // row r / 8+r, lane%16 = column), first-min tie-break on index.
#pragma unroll
    for (int off = 1; off < 16; off <<= 1) {
#pragma unroll
        for (int r = 0; r < 8; ++r) {
            float ob = __shfl_xor(best0[r], off, 32);
            int   oi = __shfl_xor(bidx0[r], off, 32);
            bool tk = (ob < best0[r]) || (ob == best0[r] && oi < bidx0[r]);
            best0[r] = tk ? ob : best0[r];
            bidx0[r] = tk ? oi : bidx0[r];
            ob = __shfl_xor(best1[r], off, 32);
            oi = __shfl_xor(bidx1[r], off, 32);
            tk = (ob < best1[r]) || (ob == best1[r] && oi < bidx1[r]);
            best1[r] = tk ? ob : best1[r];
            bidx1[r] = tk ? oi : bidx1[r];
        }
    }

    if ((lane & 15) == 0) {
        const int mhalf = (lane >> 4) * 8;
        const int base  = nchunk * MM + rowBase;
#pragma unroll
        for (int r = 0; r < 8; ++r) {
            pbest[base + mhalf + r]      = best0[r];
            pidx [base + mhalf + r]      = bidx0[r];
            pbest[base + 16 + mhalf + r] = best1[r];
            pidx [base + 16 + mhalf + r] = bidx1[r];
        }
    }
}

// ---------------------------------------------------------------------------
// Kernel 4: merge NSPLIT partial argmins (ascending chunk order keeps the
// first-min tie-break of jnp.argmin).
// ---------------------------------------------------------------------------
__global__ __launch_bounds__(256)
void vq_merge(const float* __restrict__ pbest,
              const int* __restrict__ pidx,
              int* __restrict__ mind) {
    const int r = blockIdx.x * blockDim.x + threadIdx.x;
    float b = pbest[r];
    int   ix = pidx[r];
#pragma unroll
    for (int c = 1; c < NSPLIT; ++c) {
        float ob = pbest[c * MM + r];
        int   oi = pidx[c * MM + r];
        bool tk = (ob < b) || (ob == b && oi < ix);
        b  = tk ? ob : b;
        ix = tk ? oi : ix;
    }
    mind[r] = ix;
}

// ---------------------------------------------------------------------------
// Kernel 5: gather + straight-through output + loss partial sums.
// ---------------------------------------------------------------------------
__global__ __launch_bounds__(256)
void vq_gather(const float* __restrict__ z,
               const float* __restrict__ emb,
               const int* __restrict__ mind,
               float* __restrict__ out,
               float* __restrict__ loss_acc) {
    const int i = blockIdx.x * blockDim.x + threadIdx.x;
    const int b = i >> 18;          // / CHW
    const int j = i & (CHW - 1);
    const int n = j >> 8;           // / C
    const int c = j & (CC - 1);
    const int k = mind[b * HWW + n];
    const float zq = emb[(size_t)k * CC + c];
    const float zv = z[i];
    out[i] = 2.0f * zv - zq;
    float d = zq - zv;
    float v = d * d;
#pragma unroll
    for (int off = 16; off > 0; off >>= 1)
        v += __shfl_xor(v, off, 32);
    __shared__ float sred[8];
    if ((threadIdx.x & 31) == 0) sred[threadIdx.x >> 5] = v;
    __syncthreads();
    if (threadIdx.x == 0) {
        float s = 0.0f;
#pragma unroll
        for (int w = 0; w < 8; ++w) s += sred[w];
        atomicAdd(loss_acc, s);
    }
}

// ---------------------------------------------------------------------------
// Kernel 6: finalize loss = 1.25 * mean(diff^2)
// ---------------------------------------------------------------------------
__global__ void vq_finalize(const float* __restrict__ loss_acc,
                            float* __restrict__ loss_out) {
    *loss_out = (*loss_acc) * (1.25f / (float)TOTAL);
}

// ---------------------------------------------------------------------------
extern "C" void kernel_launch(void* const* d_in, const int* in_sizes, int n_in,
                              void* d_out, int out_size, void* d_ws, size_t ws_size,
                              hipStream_t stream) {
    const float* z   = (const float*)d_in[0];
    const float* emb = (const float*)d_in[1];

    char* ws = (char*)d_ws;                 // needs ~9.5 MB
    __bf16* ehi    = (__bf16*)(ws + EHI_OFF);
    __bf16* elo    = (__bf16*)(ws + ELO_OFF);
    float*  enorm  = (float*)(ws + ENORM_OFF);
    float*  lossac = (float*)(ws + LOSS_OFF);
    float*  pbest  = (float*)(ws + PBEST_OFF);
    int*    pidx   = (int*)(ws + PIDX_OFF);

    float* outf     = (float*)d_out;
    float* zq_out   = outf;                 // TOTAL
    int*   mind     = (int*)(outf + TOTAL); // MM
    float* loss_out = outf + TOTAL + MM;    // 1

    vq_prep_split<<<(KK * CC) / 256, 256, 0, stream>>>(emb, ehi, elo, lossac);
    vq_prep_enorm<<<KK / 8, 256, 0, stream>>>(emb, enorm);
    vq_argmin<<<128 * NSPLIT, 256, 0, stream>>>(z, ehi, elo, enorm, pbest, pidx);
    vq_merge<<<MM / 256, 256, 0, stream>>>(pbest, pidx, mind);
    vq_gather<<<TOTAL / 256, 256, 0, stream>>>(z, emb, mind, zq_out, lossac);
    vq_finalize<<<1, 1, 0, stream>>>(lossac, loss_out);
}
